// BCQ_49615462203455
// MI455X (gfx1250) — matrix operand order repaired and checked by reading, not compiled, not run on hardware
//
#include <hip/hip_runtime.h>
#include <hip/hip_bf16.h>
#include <math.h>

typedef __attribute__((ext_vector_type(16))) _Float16 v16h;
typedef __attribute__((ext_vector_type(8)))  _Float16 v8h;
typedef __attribute__((ext_vector_type(8)))  float    v8f;

// ---------------------------------------------------------------------------
// Problem constants
// ---------------------------------------------------------------------------
#define BATCH      16384
#define NSAMP      10
#define NROWS      (BATCH * NSAMP)      // 163840
#define ROWS_PER_BLK 64                 // 4 WMMA row-tiles per workgroup
#define NBLOCKS    (NROWS / ROWS_PER_BLK)

// Padded layer dims (K multiple of 32, N multiple of 16)
#define DEC_K1 32   // 21 real
#define DEC_N1 768  // 750 real
#define DEC_K2 768
#define DEC_N2 768
#define DEC_K3 768
#define DEC_N3 16   // 3 real
#define Q_K1   32   // 14 real
#define Q_N1   416  // 400 real
#define Q_K2   416
#define Q_N2   320  // 300 real
#define Q_K3   320
#define Q_N3   16   // 3 or 1 real

#define ACT_NONE 0
#define ACT_RELU 1
#define ACT_TANH 2

// LDS layout in _Float16 units (total 214272 bytes < 320KB/WGP)
#define SM_XBUF   0                      // 64 x 32 f16
#define SM_SBUF   2048                   // 64 x 16 f16
#define SM_ABUF   3072                   // 64 x 16 f16
#define SM_DABUF  4096                   // 64 x 16 f16
#define SM_QTMP   5120                   // 64 f32   (128 f16 slots)
#define SM_BIAS   5248                   // 768 f32  (1536 f16 slots)
#define SM_QOUT   6784                   // 64 x 16 f32 (2048 f16 slots)
#define SM_BIG0   8832                   // 64 x 768 f16
#define SM_BIG1   57984                  // 64 x 768 f16
#define SM_TOTAL_F16 107136
#define SM_BYTES (SM_TOTAL_F16 * 2)

// ---------------------------------------------------------------------------
// Weight pre-pack: fp32 row-major [kreal x nreal] -> f16 WMMA B-fragment tiles
// Packed index p = ((kt*ntiles + nt)*32 + lane)*16 + e
//   maps element  k = kt*32 + (lane>>4)*16 + e ,  n = nt*16 + (lane&15)
// (contiguous k order used here, matched by the A loads: lanes 0-15 hold K=0..15,
//  lanes 16-31 hold K=16..31)
// ---------------------------------------------------------------------------
__global__ void bcq_pack_w(const float* __restrict__ src, _Float16* __restrict__ dst,
                           int kreal, int nreal, int Kpad, int Npad) {
    const int ntiles = Npad >> 4;
    const int total  = Kpad * Npad;
    for (int p = blockIdx.x * blockDim.x + threadIdx.x; p < total;
         p += gridDim.x * blockDim.x) {
        int e    = p & 15;
        int lane = (p >> 4) & 31;
        int tile = p >> 9;
        int nt   = tile % ntiles;
        int kt   = tile / ntiles;
        int k    = kt * 32 + (lane >> 4) * 16 + e;
        int n    = nt * 16 + (lane & 15);
        float v  = (k < kreal && n < nreal) ? src[(size_t)k * nreal + n] : 0.0f;
        dst[p] = (_Float16)v;
    }
}

// ---------------------------------------------------------------------------
// In-LDS GEMM: out[64 x Npad] = act(in[64 x Kpad] @ Wpacked + biasLDS)
// 8 waves; wave's row-tile mt = wave&3 is loop-invariant; the A operand
// is re-read from LDS for every column tile. B fragments are software
// pipelined (load kt+1 before WMMA kt) so global loads overlap the matrix op.
// ---------------------------------------------------------------------------
template <int ACT, bool OUTF32>
__device__ __forceinline__ void lds_gemm(const _Float16* __restrict__ inBuf, int istride,
                                         int Kpad,
                                         const _Float16* __restrict__ Wp,
                                         const float* __restrict__ biasLDS,
                                         int Npad,
                                         _Float16* __restrict__ outH,
                                         float* __restrict__ outF, int ostride) {
    const int lane   = threadIdx.x & 31;
    const int wave   = threadIdx.x >> 5;
    const int half   = lane >> 4;
    const int r      = lane & 15;
    const int ntiles = Npad >> 4;
    const int ksteps = Kpad >> 5;
    const int ttotal = 4 * ntiles;      // 4 row tiles x ntiles col tiles

    for (int t = wave; t < ttotal; t += 8) {
        const int mt = t & 3;
        const int nt = t >> 2;
        const int row = mt * 16 + r;
        const _Float16* arow  = inBuf + (size_t)row * istride + half * 16;
        const _Float16* wtile = Wp + ((size_t)nt * 512) + lane * 16;

        // warm next tile's first B chunk in the WGP cache (branch-free:
        // tail iterations re-prefetch the current tile, which is harmless)
        {
            int tn = (t + 8 < ttotal) ? (t + 8) : t;
            __builtin_prefetch(Wp + ((size_t)(tn >> 2) * 512) + lane * 16, 0, 3);
        }

        v8f acc = {};
        v16h bf = *(const v16h*)(wtile);
        #pragma unroll
        for (int kt = 0; kt < ksteps; ++kt) {
            v16h bcur = bf;
            if (kt + 1 < ksteps)
                bf = *(const v16h*)(wtile + (size_t)(kt + 1) * ntiles * 512);
            v8h a0 = *(const v8h*)(arow + kt * 32);
            v8h a1 = *(const v8h*)(arow + kt * 32 + 8);
            v16h af = __builtin_shufflevector(a0, a1,
                        0, 1, 2, 3, 4, 5, 6, 7, 8, 9, 10, 11, 12, 13, 14, 15);
            acc = __builtin_amdgcn_wmma_f32_16x16x32_f16(
                      false, af, false, bcur, (short)0, acc, false, false);
        }
        const int col = nt * 16 + r;
        const float bv = biasLDS[col];
        #pragma unroll
        for (int i = 0; i < 8; ++i) {
            float v = acc[i] + bv;
            if (ACT == ACT_RELU) v = v > 0.0f ? v : 0.0f;
            if (ACT == ACT_TANH) v = tanhf(v);
            if (OUTF32)
                outF[(size_t)(mt * 16 + half * 8 + i) * ostride + col] = v;
            else
                outH[(size_t)(mt * 16 + half * 8 + i) * ostride + col] = (_Float16)v;
        }
    }
}

// Stage one layer's bias (zero-padded) into LDS as f32.
__device__ __forceinline__ void stage_bias(float* __restrict__ BIASF,
                                           const float* __restrict__ bias,
                                           int nreal, int Npad, int tid) {
    for (int i = tid; i < Npad; i += 256)
        BIASF[i] = (i < nreal) ? bias[i] : 0.0f;
}

// ---------------------------------------------------------------------------
// Fused BCQ pipeline: each block handles 64 consecutive (state,sample) rows.
// ---------------------------------------------------------------------------
struct BcqArgs {
    const float* states;  const float* noise;
    const _Float16* dec1p; const float* dec_b1;
    const _Float16* dec2p; const float* dec_b2;
    const _Float16* dec3p; const float* dec_b3;
    const _Float16* act1p; const float* act_b1;
    const _Float16* act2p; const float* act_b2;
    const _Float16* act3p; const float* act_b3;
    const _Float16* q11p;  const float* q1_b1;
    const _Float16* q12p;  const float* q1_b2;
    const _Float16* q13p;  const float* q1_b3;
    const _Float16* q21p;  const float* q2_b1;
    const _Float16* q22p;  const float* q2_b2;
    const _Float16* q23p;  const float* q2_b3;
    _Float16* actions_g;   float* qmin_g;
};

__global__ __launch_bounds__(256) void bcq_main(BcqArgs A) {
    extern __shared__ _Float16 smem[];
    _Float16* XBUF  = smem + SM_XBUF;
    _Float16* SBUF  = smem + SM_SBUF;
    _Float16* ABUF  = smem + SM_ABUF;
    _Float16* DABUF = smem + SM_DABUF;
    float*    QTMP  = (float*)(smem + SM_QTMP);
    float*    BIASF = (float*)(smem + SM_BIAS);
    float*    QOUT  = (float*)(smem + SM_QOUT);
    _Float16* BIG0  = smem + SM_BIG0;
    _Float16* BIG1  = smem + SM_BIG1;

    const int tid = threadIdx.x;
    const int blockBase = blockIdx.x * ROWS_PER_BLK;

    // --- Stage 0: X0 = [s(11), clip(z,-.5,.5)(10), 0pad] ; stash s in SBUF ---
    for (int idx = tid; idx < ROWS_PER_BLK * 32; idx += 256) {
        int row = idx >> 5, col = idx & 31;
        int grow = blockBase + row;
        float v = 0.0f;
        if (col < 11)       v = A.states[(size_t)(grow / NSAMP) * 11 + col];
        else if (col < 21)  {
            float z = A.noise[(size_t)grow * 10 + (col - 11)];
            v = fminf(fmaxf(z, -0.5f), 0.5f);
        }
        XBUF[idx] = (_Float16)v;
    }
    for (int idx = tid; idx < ROWS_PER_BLK * 16; idx += 256) {
        int row = idx >> 4, col = idx & 15;
        float v = (col < 11) ? A.states[(size_t)((blockBase + row) / NSAMP) * 11 + col] : 0.0f;
        SBUF[idx] = (_Float16)v;
    }

    // --- VAE decoder ---
    stage_bias(BIASF, A.dec_b1, 750, DEC_N1, tid);
    __syncthreads();
    lds_gemm<ACT_RELU, false>(XBUF, 32, DEC_K1, A.dec1p, BIASF, DEC_N1, BIG0, nullptr, DEC_N1);
    __syncthreads();
    stage_bias(BIASF, A.dec_b2, 750, DEC_N2, tid);
    __syncthreads();
    lds_gemm<ACT_RELU, false>(BIG0, DEC_N1, DEC_K2, A.dec2p, BIASF, DEC_N2, BIG1, nullptr, DEC_N2);
    __syncthreads();
    stage_bias(BIASF, A.dec_b3, 3, DEC_N3, tid);
    __syncthreads();
    lds_gemm<ACT_TANH, false>(BIG1, DEC_N2, DEC_K3, A.dec3p, BIASF, DEC_N3, ABUF, nullptr, 16);
    __syncthreads();

    // --- X1 = [s, a_vae] ---
    for (int idx = tid; idx < ROWS_PER_BLK * 32; idx += 256) {
        int row = idx >> 5, col = idx & 31;
        _Float16 v = (_Float16)0.0f;
        if (col < 11)       v = SBUF[row * 16 + col];
        else if (col < 14)  v = ABUF[row * 16 + (col - 11)];
        XBUF[idx] = v;
    }
    stage_bias(BIASF, A.act_b1, 400, Q_N1, tid);
    __syncthreads();

    // --- Perturbation actor ---
    lds_gemm<ACT_RELU, false>(XBUF, 32, Q_K1, A.act1p, BIASF, Q_N1, BIG0, nullptr, Q_N1);
    __syncthreads();
    stage_bias(BIASF, A.act_b2, 300, Q_N2, tid);
    __syncthreads();
    lds_gemm<ACT_RELU, false>(BIG0, Q_N1, Q_K2, A.act2p, BIASF, Q_N2, BIG1, nullptr, Q_N2);
    __syncthreads();
    stage_bias(BIASF, A.act_b3, 3, Q_N3, tid);
    __syncthreads();
    lds_gemm<ACT_TANH, false>(BIG1, Q_N2, Q_K3, A.act3p, BIASF, Q_N3, DABUF, nullptr, 16);
    __syncthreads();

    // --- actions = clamp(0.05*da + a, -1, 1) (overwrite ABUF) ---
    {
        int row = tid >> 2, col = tid & 3;
        if (col < 3) {
            float a  = (float)ABUF[row * 16 + col];
            float da = (float)DABUF[row * 16 + col];
            float v  = fminf(fmaxf(0.05f * da + a, -1.0f), 1.0f);
            ABUF[row * 16 + col] = (_Float16)v;
            A.actions_g[(size_t)(blockBase + row) * 4 + col] = (_Float16)v;
        }
    }
    __syncthreads();

    // --- X2 = [s, actions] ---
    for (int idx = tid; idx < ROWS_PER_BLK * 32; idx += 256) {
        int row = idx >> 5, col = idx & 31;
        _Float16 v = (_Float16)0.0f;
        if (col < 11)       v = SBUF[row * 16 + col];
        else if (col < 14)  v = ABUF[row * 16 + (col - 11)];
        XBUF[idx] = v;
    }
    stage_bias(BIASF, A.q1_b1, 400, Q_N1, tid);
    __syncthreads();

    // --- Critic 1 ---
    lds_gemm<ACT_RELU, false>(XBUF, 32, Q_K1, A.q11p, BIASF, Q_N1, BIG0, nullptr, Q_N1);
    __syncthreads();
    stage_bias(BIASF, A.q1_b2, 300, Q_N2, tid);
    __syncthreads();
    lds_gemm<ACT_RELU, false>(BIG0, Q_N1, Q_K2, A.q12p, BIASF, Q_N2, BIG1, nullptr, Q_N2);
    __syncthreads();
    stage_bias(BIASF, A.q1_b3, 1, Q_N3, tid);
    __syncthreads();
    lds_gemm<ACT_NONE, true>(BIG1, Q_N2, Q_K3, A.q13p, BIASF, Q_N3, nullptr, QOUT, 16);
    __syncthreads();
    if (tid < ROWS_PER_BLK) QTMP[tid] = QOUT[tid * 16];
    stage_bias(BIASF, A.q2_b1, 400, Q_N1, tid);
    __syncthreads();

    // --- Critic 2 ---
    lds_gemm<ACT_RELU, false>(XBUF, 32, Q_K1, A.q21p, BIASF, Q_N1, BIG0, nullptr, Q_N1);
    __syncthreads();
    stage_bias(BIASF, A.q2_b2, 300, Q_N2, tid);
    __syncthreads();
    lds_gemm<ACT_RELU, false>(BIG0, Q_N1, Q_K2, A.q22p, BIASF, Q_N2, BIG1, nullptr, Q_N2);
    __syncthreads();
    stage_bias(BIASF, A.q2_b3, 1, Q_N3, tid);
    __syncthreads();
    lds_gemm<ACT_NONE, true>(BIG1, Q_N2, Q_K3, A.q23p, BIASF, Q_N3, nullptr, QOUT, 16);
    __syncthreads();
    if (tid < ROWS_PER_BLK) {
        A.qmin_g[blockBase + tid] = fminf(QTMP[tid], QOUT[tid * 16]);
    }
}

// ---------------------------------------------------------------------------
// Per-state argmax over 10 samples + gather winning action (fp32 out)
// ---------------------------------------------------------------------------
__global__ void bcq_select(const float* __restrict__ qmin,
                           const _Float16* __restrict__ acts,
                           float* __restrict__ out) {
    int i = blockIdx.x * blockDim.x + threadIdx.x;
    if (i >= BATCH) return;
    int base = i * NSAMP;
    float best = qmin[base];
    int bj = 0;
    for (int j = 1; j < NSAMP; ++j) {
        float q = qmin[base + j];
        if (q > best) { best = q; bj = j; }   // first-max = jnp.argmax semantics
    }
    const _Float16* ap = acts + (size_t)(base + bj) * 4;
    out[(size_t)i * 3 + 0] = (float)ap[0];
    out[(size_t)i * 3 + 1] = (float)ap[1];
    out[(size_t)i * 3 + 2] = (float)ap[2];
}

// ---------------------------------------------------------------------------
extern "C" void kernel_launch(void* const* d_in, const int* in_sizes, int n_in,
                              void* d_out, int out_size, void* d_ws, size_t ws_size,
                              hipStream_t stream) {
    const float* states = (const float*)d_in[0];
    const float* noise  = (const float*)d_in[1];
    const float* w[12]  = { (const float*)d_in[2],  (const float*)d_in[4],
                            (const float*)d_in[6],  (const float*)d_in[8],
                            (const float*)d_in[10], (const float*)d_in[12],
                            (const float*)d_in[14], (const float*)d_in[16],
                            (const float*)d_in[18], (const float*)d_in[20],
                            (const float*)d_in[22], (const float*)d_in[24] };
    const int kreal[12] = { 21, 750, 750, 14, 400, 300, 14, 400, 300, 14, 400, 300 };
    const int nreal[12] = { 750, 750, 3,  400, 300, 3,  400, 300, 1,  400, 300, 1 };
    const int Kpad[12]  = { DEC_K1, DEC_K2, DEC_K3, Q_K1, Q_K2, Q_K3,
                            Q_K1, Q_K2, Q_K3, Q_K1, Q_K2, Q_K3 };
    const int Npad[12]  = { DEC_N1, DEC_N2, DEC_N3, Q_N1, Q_N2, Q_N3,
                            Q_N1, Q_N2, Q_N3, Q_N1, Q_N2, Q_N3 };

    char* ws = (char*)d_ws;
    size_t off = 0;
    auto alloc = [&](size_t bytes) -> void* {
        off = (off + 255) & ~(size_t)255;
        void* p = ws + off;
        off += bytes;
        return p;
    };

    _Float16* actions_g = (_Float16*)alloc((size_t)NROWS * 4 * sizeof(_Float16));
    float*    qmin_g    = (float*)   alloc((size_t)NROWS * sizeof(float));

    _Float16* wp[12];
    for (int m = 0; m < 12; ++m) {
        wp[m] = (_Float16*)alloc((size_t)Kpad[m] * Npad[m] * sizeof(_Float16));
        int total = Kpad[m] * Npad[m];
        int blocks = (total + 255) / 256;
        bcq_pack_w<<<blocks, 256, 0, stream>>>(w[m], wp[m], kreal[m], nreal[m],
                                               Kpad[m], Npad[m]);
    }

    BcqArgs A;
    A.states = states;  A.noise = noise;
    A.dec1p = wp[0];  A.dec_b1 = (const float*)d_in[3];
    A.dec2p = wp[1];  A.dec_b2 = (const float*)d_in[5];
    A.dec3p = wp[2];  A.dec_b3 = (const float*)d_in[7];
    A.act1p = wp[3];  A.act_b1 = (const float*)d_in[9];
    A.act2p = wp[4];  A.act_b2 = (const float*)d_in[11];
    A.act3p = wp[5];  A.act_b3 = (const float*)d_in[13];
    A.q11p  = wp[6];  A.q1_b1  = (const float*)d_in[15];
    A.q12p  = wp[7];  A.q1_b2  = (const float*)d_in[17];
    A.q13p  = wp[8];  A.q1_b3  = (const float*)d_in[19];
    A.q21p  = wp[9];  A.q2_b1  = (const float*)d_in[21];
    A.q22p  = wp[10]; A.q2_b2  = (const float*)d_in[23];
    A.q23p  = wp[11]; A.q2_b3  = (const float*)d_in[25];
    A.actions_g = actions_g;
    A.qmin_g    = qmin_g;

    bcq_main<<<NBLOCKS, 256, SM_BYTES, stream>>>(A);

    bcq_select<<<(BATCH + 255) / 256, 256, 0, stream>>>(qmin_g, actions_g,
                                                        (float*)d_out);
}
